// ShiftedPatchTokenization_13769665151173
// MI455X (gfx1250) — matrix operand order repaired and checked by reading, not compiled, not run on hardware
//
#include <hip/hip_runtime.h>
#include <hip/hip_bf16.h>

// ---------------------------------------------------------------------------
// CDNA5 / gfx1250 Shifted Patch Tokenization
//   out = LN1(patchify(out5,16)) @ W1 + b1 + LN2(patchify(haar(out5),8)) @ W2 + b2
// bf16 WMMA (v_wmma_f32_16x16x32_bf16), fp32 LayerNorm + accumulation.
// GEMM: 32x64 register-blocked tile per wave32 (8 accumulators) -> 1.5 b128
// loads per WMMA (2 A-frags + 4 B-frags per K32 step feed 8 WMMAs).
// ---------------------------------------------------------------------------

typedef __bf16 v16bf __attribute__((ext_vector_type(16)));
typedef float  v8f   __attribute__((ext_vector_type(8)));
typedef unsigned int u32x4 __attribute__((ext_vector_type(4)));

#define BATCH  64
#define GRID_HW 14
#define TOKENS (BATCH * GRID_HW * GRID_HW)   // 12544
#define DIM    768
#define K1     3840                          // 16*16*15
#define K2     960                           // 8*8*15
#define NK1    (K1 / 32)                     // 120 wmma K-steps
#define NK2    (K2 / 32)                     // 30
#define NT     (DIM / 16)                    // 48 N 16-tiles
#define MT32   (TOKENS / 32)                 // 392 M 32-tiles
#define NT64   (DIM / 64)                    // 12  N 64-tiles

__device__ __forceinline__ unsigned short f2bf(float f) {
    // round-to-nearest-even fp32 -> bf16
    unsigned u = __builtin_bit_cast(unsigned, f);
    u += 0x7FFFu + ((u >> 16) & 1u);
    return (unsigned short)(u >> 16);
}

// ---------------------------------------------------------------------------
// Repack a (K x DIM) row-major fp32 weight matrix into per-lane WMMA B
// fragments (bf16).  Fragment layout per ISA 7.12.2 (16-bit B, 32x16):
//   lane holds N = nTile*16 + (lane&15), K = kStep*32 + 8*(lane>>4) + s (+8 for s>=8)
// Stored as wp[((nTile*nK + kStep)*32 + lane)*16 + s]  -> one 32B load/lane.
// ---------------------------------------------------------------------------
__global__ __launch_bounds__(256) void pack_w_kernel(const float* __restrict__ w,
                                                     unsigned short* __restrict__ wp,
                                                     int nK) {
    const int id    = blockIdx.x * 256 + threadIdx.x;   // nT*nK*32 threads
    const int lane  = id & 31;
    const int grp   = id >> 5;
    const int kStep = grp % nK;
    const int nTile = grp / nK;
    const int n     = nTile * 16 + (lane & 15);
    const int kbase = kStep * 32 + ((lane >> 4) << 3);

    alignas(16) unsigned short tmp[16];
#pragma unroll
    for (int s = 0; s < 16; ++s)
        tmp[s] = f2bf(w[(size_t)(kbase + s + ((s >> 3) << 3)) * DIM + n]);   // coalesced across lanes

    u32x4* outp = (u32x4*)(wp + (size_t)id * 16);
    outp[0] = *(const u32x4*)(tmp);
    outp[1] = *(const u32x4*)(tmp + 8);
}

// ---------------------------------------------------------------------------
// One workgroup per token: build shifted 5-channel-group patch in LDS,
// Haar detail means, two LayerNorms, emit row-major bf16 A1 (3840) / A2 (960).
// ---------------------------------------------------------------------------
__global__ __launch_bounds__(256) void pack_a_kernel(
    const float* __restrict__ x,
    const float* __restrict__ g1, const float* __restrict__ be1,
    const float* __restrict__ g2, const float* __restrict__ be2,
    unsigned short* __restrict__ A1, unsigned short* __restrict__ A2)
{
    __shared__ float p5[K1];        // patch values, index = (ph*16+pw)*15 + cc
    __shared__ float wv[K2];        // haar values,  index = (ph*8+pw)*15 + cc
    __shared__ float rs[256], rq[256];
    __shared__ float s_mean, s_rstd;

    const int t     = threadIdx.x;
    const int token = blockIdx.x;
    const int b     = token / 196;
    const int rr    = token % 196;
    const int gh    = rr / GRID_HW, gw = rr % GRID_HW;
    const int hbase = gh * 16 + (t >> 4);
    const int wbase = gw * 16 + (t & 15);

    const int dy[5] = {0, -8, -8,  8,  8};
    const int dx[5] = {0, -8,  8, -8,  8};

    // --- gather shifted crops: thread t = in-patch position, loop cc ---
#pragma unroll
    for (int cc = 0; cc < 15; ++cc) {
        const int si = cc / 3, c = cc % 3;
        const int h = hbase + dy[si];
        const int w = wbase + dx[si];
        float v = 0.0f;
        if ((unsigned)h < 224u && (unsigned)w < 224u)
            v = x[(((size_t)b * 3 + c) * 224 + h) * 224 + w];
        p5[t * 15 + cc] = v;        // stride-15 -> bank-conflict-free
    }
    __syncthreads();

    // --- Haar detail mean (all operands live in this patch's LDS) ---
    for (int j = t; j < K2; j += 256) {
        const int pos2 = j / 15, cc = j % 15;
        const int base = (((pos2 >> 3) * 2) * 16 + (pos2 & 7) * 2) * 15 + cc;
        const float a  = p5[base];
        const float bq = p5[base + 15];
        const float cq = p5[base + 240];
        const float dq = p5[base + 255];
        wv[j] = (3.0f * a - bq - cq - dq) * (1.0f / 6.0f);
    }

    // --- LayerNorm 1 over 3840 ---
    float sm = 0.f, sq = 0.f;
#pragma unroll
    for (int q = 0; q < 15; ++q) { const float v = p5[t * 15 + q]; sm += v; sq += v * v; }
    rs[t] = sm; rq[t] = sq;
    __syncthreads();
    for (int s = 128; s > 0; s >>= 1) {
        if (t < s) { rs[t] += rs[t + s]; rq[t] += rq[t + s]; }
        __syncthreads();
    }
    if (t == 0) {
        const float m = rs[0] * (1.0f / K1);
        s_mean = m;
        s_rstd = rsqrtf(rq[0] * (1.0f / K1) - m * m + 1e-5f);
    }
    __syncthreads();
    {
        const float m1 = s_mean, r1 = s_rstd;
        unsigned short* outr = A1 + (size_t)token * K1;
#pragma unroll
        for (int q = 0; q < 15; ++q) {
            const int j = t * 15 + q;
            outr[j] = f2bf((p5[j] - m1) * r1 * g1[j] + be1[j]);
        }
    }

    // --- LayerNorm 2 over 960 ---
    sm = 0.f; sq = 0.f;
    for (int j = t; j < K2; j += 256) { const float v = wv[j]; sm += v; sq += v * v; }
    rs[t] = sm; rq[t] = sq;
    __syncthreads();
    for (int s = 128; s > 0; s >>= 1) {
        if (t < s) { rs[t] += rs[t + s]; rq[t] += rq[t + s]; }
        __syncthreads();
    }
    if (t == 0) {
        const float m = rs[0] * (1.0f / K2);
        s_mean = m;
        s_rstd = rsqrtf(rq[0] * (1.0f / K2) - m * m + 1e-5f);
    }
    __syncthreads();
    {
        const float m2 = s_mean, r2 = s_rstd;
        unsigned short* outr = A2 + (size_t)token * K2;
        for (int j = t; j < K2; j += 256)
            outr[j] = f2bf((wv[j] - m2) * r2 * g2[j] + be2[j]);
    }
}

// ---------------------------------------------------------------------------
// Fused GEMM, 32x64 tile per wave32 (8 f32 accumulators):
//   acc(m,n) = b1[n] + b2[n] + sum_k A1*W1 + sum_k A2*W2
// A fragment per ISA: lane M = lane&15; K runs [hi*8,hi*8+8) and +16 -> two
// b128 loads from row-major bf16.  B fragments pre-packed (2x b128 per lane).
// Per K32 step: 12 b128 loads feed 8 independent WMMAs (1.5 loads/WMMA).
// ---------------------------------------------------------------------------
__global__ __launch_bounds__(256) void gemm_kernel(
    const unsigned short* __restrict__ A1, const unsigned short* __restrict__ A2,
    const unsigned short* __restrict__ W1, const unsigned short* __restrict__ W2,
    const float* __restrict__ b1, const float* __restrict__ b2,
    float* __restrict__ out)
{
    const int lane  = threadIdx.x & 31;
    const int wave  = threadIdx.x >> 5;
    const int tile  = blockIdx.x * 8 + wave;   // MT32*NT64 = 4704 tiles, exact
    const int m32   = tile / NT64;
    const int n64   = tile % NT64;
    const int ln    = lane & 15;
    const int hi    = lane >> 4;
    const int n0    = n64 * 64 + ln;           // columns n0, n0+16, n0+32, n0+48

    v8f acc[2][4];                             // [m sub][n sub]
#pragma unroll
    for (int j = 0; j < 4; ++j) {
        const float bias = b1[n0 + j * 16] + b2[n0 + j * 16];
#pragma unroll
        for (int r = 0; r < 8; ++r) { acc[0][j][r] = bias; acc[1][j][r] = bias; }
    }

    union frag { v16bf v; u32x4 q[2]; };

    // ---- GEMM 1: K = 3840 ----
    {
        const unsigned short* ap0 = A1 + (size_t)(m32 * 32 + ln) * K1 + hi * 8;
        const unsigned short* ap1 = ap0 + (size_t)16 * K1;
        const unsigned short* wp  = W1 + ((size_t)(n64 * 4) * NK1 * 32 + lane) * 16;
#pragma unroll 2
        for (int k = 0; k < NK1; ++k) {
            __builtin_prefetch((const void*)(ap0 + (k + 16) * 32), 0, 3);
            __builtin_prefetch((const void*)(ap1 + (k + 16) * 32), 0, 3);
            frag a0, a1f;
            a0.q[0]  = *(const u32x4*)(ap0 + k * 32);
            a0.q[1]  = *(const u32x4*)(ap0 + k * 32 + 16);
            a1f.q[0] = *(const u32x4*)(ap1 + k * 32);
            a1f.q[1] = *(const u32x4*)(ap1 + k * 32 + 16);
#pragma unroll
            for (int j = 0; j < 4; ++j) {
                frag bf;
                const unsigned short* wpj = wp + (size_t)j * NK1 * 512 + (size_t)k * 512;
                bf.q[0] = *(const u32x4*)(wpj);
                bf.q[1] = *(const u32x4*)(wpj + 8);
                acc[0][j] = __builtin_amdgcn_wmma_f32_16x16x32_bf16(
                    false, a0.v,  false, bf.v, (short)0, acc[0][j], false, false);
                acc[1][j] = __builtin_amdgcn_wmma_f32_16x16x32_bf16(
                    false, a1f.v, false, bf.v, (short)0, acc[1][j], false, false);
            }
        }
    }

    // ---- GEMM 2: K = 960 ----
    {
        const unsigned short* ap0 = A2 + (size_t)(m32 * 32 + ln) * K2 + hi * 8;
        const unsigned short* ap1 = ap0 + (size_t)16 * K2;
        const unsigned short* wp  = W2 + ((size_t)(n64 * 4) * NK2 * 32 + lane) * 16;
#pragma unroll 2
        for (int k = 0; k < NK2; ++k) {
            frag a0, a1f;
            a0.q[0]  = *(const u32x4*)(ap0 + k * 32);
            a0.q[1]  = *(const u32x4*)(ap0 + k * 32 + 16);
            a1f.q[0] = *(const u32x4*)(ap1 + k * 32);
            a1f.q[1] = *(const u32x4*)(ap1 + k * 32 + 16);
#pragma unroll
            for (int j = 0; j < 4; ++j) {
                frag bf;
                const unsigned short* wpj = wp + (size_t)j * NK2 * 512 + (size_t)k * 512;
                bf.q[0] = *(const u32x4*)(wpj);
                bf.q[1] = *(const u32x4*)(wpj + 8);
                acc[0][j] = __builtin_amdgcn_wmma_f32_16x16x32_bf16(
                    false, a0.v,  false, bf.v, (short)0, acc[0][j], false, false);
                acc[1][j] = __builtin_amdgcn_wmma_f32_16x16x32_bf16(
                    false, a1f.v, false, bf.v, (short)0, acc[1][j], false, false);
            }
        }
    }

    // D layout: lanes0-15 -> M = vgpr, lanes16-31 -> M = 8+vgpr; N = lane&15
    {
        float* orow0 = out + (size_t)(m32 * 32 + hi * 8) * DIM;       // m sub-tile 0
        float* orow1 = orow0 + (size_t)16 * DIM;                      // m sub-tile 1
#pragma unroll
        for (int r = 0; r < 8; ++r) {
#pragma unroll
            for (int j = 0; j < 4; ++j) {
                orow0[(size_t)r * DIM + n0 + j * 16] = acc[0][j][r];
                orow1[(size_t)r * DIM + n0 + j * 16] = acc[1][j][r];
            }
        }
    }
}

// ---------------------------------------------------------------------------
extern "C" void kernel_launch(void* const* d_in, const int* in_sizes, int n_in,
                              void* d_out, int out_size, void* d_ws, size_t ws_size,
                              hipStream_t stream) {
    const float* x   = (const float*)d_in[0];
    const float* g1  = (const float*)d_in[1];
    const float* be1 = (const float*)d_in[2];
    const float* w1  = (const float*)d_in[3];
    const float* b1  = (const float*)d_in[4];
    const float* g2  = (const float*)d_in[5];
    const float* be2 = (const float*)d_in[6];
    const float* w2  = (const float*)d_in[7];
    const float* b2  = (const float*)d_in[8];

    // workspace layout (bf16 ushorts), all 16B aligned:
    unsigned short* A1 = (unsigned short*)d_ws;                 // 12544*3840
    unsigned short* A2 = A1 + (size_t)TOKENS * K1;              // 12544*960
    unsigned short* W1 = A2 + (size_t)TOKENS * K2;              // 3840*768
    unsigned short* W2 = W1 + (size_t)K1 * DIM;                 // 960*768

    pack_w_kernel<<<(NT * NK1 * 32) / 256, 256, 0, stream>>>(w1, W1, NK1);
    pack_w_kernel<<<(NT * NK2 * 32) / 256, 256, 0, stream>>>(w2, W2, NK2);
    pack_a_kernel<<<TOKENS, 256, 0, stream>>>(x, g1, be1, g2, be2, A1, A2);
    gemm_kernel<<<(MT32 * NT64) / 8, 256, 0, stream>>>(
        A1, A2, W1, W2, b1, b2, (float*)d_out);
}
